// GPT2Attention_69466801045845
// MI455X (gfx1250) — matrix operand order repaired and checked by reading, not compiled, not run on hardware
//
#include <hip/hip_runtime.h>

typedef __attribute__((ext_vector_type(16))) _Float16 v16h;
typedef __attribute__((ext_vector_type(8)))  _Float16 v8h;
typedef __attribute__((ext_vector_type(8)))  float    v8f;
typedef __attribute__((ext_vector_type(4)))  unsigned int u32x4;
typedef __attribute__((ext_vector_type(8)))  int          i32x8;
typedef __attribute__((ext_vector_type(4)))  int          i32x4;

#define D_EMBED 1024
#define NHEAD   16
#define HD      64
#define BATCH   2
#define SEQ     2048
#define MTOT    (BATCH * SEQ)   // 4096 tokens
#define KDIM    1024

__device__ __forceinline__ v16h cat8(v8h lo, v8h hi) {
  v16h r;
#pragma unroll
  for (int i = 0; i < 8; ++i) { r[i] = lo[i]; r[i + 8] = hi[i]; }
  return r;
}

__device__ __forceinline__ v8f wmma16(v16h a, v16h b, v8f c) {
  return __builtin_amdgcn_wmma_f32_16x16x32_f16(
      false, a, false, b, (short)0, c, false, false);
}

__device__ __forceinline__ unsigned lds_off(const void* p) {
  // generic shared-aperture pointer: low 32 bits are the LDS byte offset
  return (unsigned)(uintptr_t)p;
}

// ---------------------------------------------------------------------------
// Tensor Data Mover: 2D f16 tile (tile_d0 elems/row, tile_d1 rows, row stride
// stride0 elems) global -> LDS, with LDS row padding (D# per ISA ch.8).
// pad_interval: rows of 2^(pi+1) DWORDs; pad_amount: (pa+1) DWORDs appended.
// ---------------------------------------------------------------------------
__device__ __forceinline__ void tdm_load_2d_f16(
    const void* gaddr, unsigned lds_byte_off,
    unsigned tile_d0, unsigned tile_d1, unsigned long long stride0,
    unsigned pad_interval, unsigned pad_amount) {
  union { u32x4 v; unsigned long long q[2]; } g0;
  union { i32x8 v; unsigned long long q[4]; } g1;
  unsigned long long ga = (unsigned long long)(uintptr_t)gaddr;
  // group0: count=1 | lds_addr[63:32] ; global_addr[120:64] | type=2 [127:126]
  g0.q[0] = 1ull | ((unsigned long long)lds_byte_off << 32);
  g0.q[1] = (ga & ((1ull << 57) - 1)) | (2ull << 62);
  // group1 word0: data_size=1(2B)<<16 | pad_enable<<20 | pi<<22 | pa<<25
  unsigned long long w0 = (1ull << 16) | (1ull << 20) |
                          ((unsigned long long)pad_interval << 22) |
                          ((unsigned long long)pad_amount << 25);
  unsigned long long td0 = tile_d0, td1 = tile_d1;   // tensor dims == tile dims
  g1.q[0] = w0 | (td0 << 48);                          // tensor_dim0 [79:48]
  g1.q[1] = (td0 >> 16) | (td1 << 16) |                // tensor_dim1 [111:80]
            ((unsigned long long)tile_d0 << 48);       // tile_dim0   [127:112]
  g1.q[2] = td1 |                                      // tile_dim1   [143:128]
            ((stride0 & 0xFFFFFFFFull) << 32);         // stride0 lo  [191:160]
  g1.q[3] = (stride0 >> 32);                           // stride0 hi  [207:192]
  i32x4 z4 = {};
#if __clang_major__ >= 23
  i32x8 z8 = {};
  __builtin_amdgcn_tensor_load_to_lds(g0.v, g1.v, z4, z4, z8, 0);
#else
  __builtin_amdgcn_tensor_load_to_lds(g0.v, g1.v, z4, z4, 0);
#endif
}

// ---------------------------------------------------------------------------
// Preconversion: x -> f16 (vectorized), weights -> f16 transposed [N][K]
// ---------------------------------------------------------------------------
__global__ __launch_bounds__(256) void cvt_f16(const float* __restrict__ in,
                                               _Float16* __restrict__ out) {
  int i = blockIdx.x * 256 + threadIdx.x;   // one float4 per thread
  float4 v = *(const float4*)(in + (size_t)i * 4);
  _Float16* d = out + (size_t)i * 4;
  d[0] = (_Float16)v.x; d[1] = (_Float16)v.y;
  d[2] = (_Float16)v.z; d[3] = (_Float16)v.w;
}

__global__ __launch_bounds__(256) void transpose_cvt(
    const float* __restrict__ in,    // [K][N]
    _Float16* __restrict__ out,      // [N][K] f16
    int K, int N) {
  __shared__ _Float16 tile[32][33];
  const int bx = blockIdx.x * 32;    // n
  const int by = blockIdx.y * 32;    // k
  const int tx = threadIdx.x & 31, ty = threadIdx.x >> 5;   // 32 x 8
#pragma unroll
  for (int i = 0; i < 32; i += 8)
    tile[ty + i][tx] = (_Float16)in[(size_t)(by + ty + i) * N + bx + tx];
  __syncthreads();
#pragma unroll
  for (int i = 0; i < 32; i += 8)
    out[(size_t)(bx + ty + i) * K + by + tx] = tile[tx][ty + i];
}

// ---------------------------------------------------------------------------
// WMMA GEMM, TDM double-buffered tiles.  C[4096,N] = A[4096,1024] x Wt^T + b
// A f16 row-major, Wt f16 pre-transposed [N][1024].
// Block tile 64x128, BK=32, 8 waves, wave tile 32x32 (2x2 WMMA).
// MODE 0: epilogue scatters f16 Q/K/Vt head-major; MODE 1: f32 C out.
// ---------------------------------------------------------------------------
template <int MODE>
__global__ __launch_bounds__(256) void gemm_wmma(
    const _Float16* __restrict__ Ain, const _Float16* __restrict__ Wt,
    const float* __restrict__ bias, int N,
    _Float16* __restrict__ Qh, _Float16* __restrict__ Kh,
    _Float16* __restrict__ Vt, float* __restrict__ Cout) {
  constexpr int BM = 64, BN = 128, BK = 32, LD = 40;  // 80B padded rows
  __shared__ __align__(16) _Float16 As[2][BM][LD];
  __shared__ __align__(16) _Float16 Bs[2][BN][LD];
  const int tid = threadIdx.x;
  const int m0 = blockIdx.x * BM, n0 = blockIdx.y * BN;
  const int wave = tid >> 5, lane = tid & 31;
  const int wm = wave & 1, wn = wave >> 1;            // 2 x 4 wave grid
  const int half = lane >> 4, l16 = lane & 15;
  const int kb = half * 8;

  v8f acc[2][2] = {};

  if (wave == 0) {   // one DMA per workgroup (TDM is a wave-level op)
    tdm_load_2d_f16(Ain + (size_t)m0 * KDIM, lds_off(&As[0][0][0]), 32, BM, KDIM, 3, 3);
    tdm_load_2d_f16(Wt  + (size_t)n0 * KDIM, lds_off(&Bs[0][0][0]), 32, BN, KDIM, 3, 3);
  }
  int t = 0;
  for (int kt = 0; kt < KDIM; kt += BK) {
    if (wave == 0) __builtin_amdgcn_s_wait_tensorcnt(0);
    __syncthreads();                    // tiles[t] ready for everyone
    if (wave == 0 && kt + BK < KDIM) {  // prefetch next tile, overlap compute
      tdm_load_2d_f16(Ain + (size_t)m0 * KDIM + kt + BK,
                      lds_off(&As[t ^ 1][0][0]), 32, BM, KDIM, 3, 3);
      tdm_load_2d_f16(Wt + (size_t)n0 * KDIM + kt + BK,
                      lds_off(&Bs[t ^ 1][0][0]), 32, BN, KDIM, 3, 3);
    }
    v16h a[2], b[2];
#pragma unroll
    for (int mt = 0; mt < 2; ++mt) {
      int r = wm * 32 + mt * 16 + l16;
      a[mt] = cat8(*(const v8h*)&As[t][r][kb], *(const v8h*)&As[t][r][kb + 16]);
    }
#pragma unroll
    for (int nt = 0; nt < 2; ++nt) {
      int c = wn * 32 + nt * 16 + l16;
      int k0 = half * 8;
      b[nt] = cat8(*(const v8h*)&Bs[t][c][k0], *(const v8h*)&Bs[t][c][k0 + 16]);
    }
#pragma unroll
    for (int mt = 0; mt < 2; ++mt)
#pragma unroll
      for (int nt = 0; nt < 2; ++nt)
        acc[mt][nt] = wmma16(a[mt], b[nt], acc[mt][nt]);
    __syncthreads();                    // done reading tiles[t]
    t ^= 1;
  }

#pragma unroll
  for (int mt = 0; mt < 2; ++mt)
#pragma unroll
    for (int nt = 0; nt < 2; ++nt)
#pragma unroll
      for (int r = 0; r < 8; ++r) {
        int grow = m0 + wm * 32 + mt * 16 + r + 8 * half;
        int gcol = n0 + wn * 32 + nt * 16 + l16;
        float v = acc[mt][nt][r] + bias[gcol];
        if (MODE == 0) {
          int bb = grow >> 11, l = grow & 2047;
          int seg = gcol >> 10, cc = gcol & 1023;
          int h = cc >> 6, d = cc & 63;
          size_t base = ((size_t)(bb * NHEAD + h) * SEQ + l) * HD + d;
          if (seg == 0)       Qh[base] = (_Float16)(v * 0.125f);  // 1/sqrt(hd)
          else if (seg == 1)  Kh[base] = (_Float16)v;
          else Vt[((size_t)(bb * NHEAD + h) * HD + d) * SEQ + l] = (_Float16)v;
        } else {
          Cout[(size_t)grow * N + gcol] = v;
        }
      }
}

// ---------------------------------------------------------------------------
// Flash attention: one wave owns 16 query rows of one (b,h).
// K/V 32-key chunks TDM-staged into LDS (double buffered, single-wave
// TENSORcnt pipelining).  S = Q K^T (4 WMMA), causal mask, online softmax
// (max via shfl_xor butterflies; row-sum via WMMA with an all-ones B),
// O += P V (4 WMMA).
// ---------------------------------------------------------------------------
__global__ __launch_bounds__(32) void attn_fa(
    const _Float16* __restrict__ Qh, const _Float16* __restrict__ Kh,
    const _Float16* __restrict__ Vt, _Float16* __restrict__ Aout) {
  __shared__ __align__(16) _Float16 Ks[2][32][72];  // 144B rows, conflict-free
  __shared__ __align__(16) _Float16 Vs[2][64][40];  // 80B rows
  __shared__ __align__(16) _Float16 P[16][40];
  const int lane = threadIdx.x;
  const int half = lane >> 4, l16 = lane & 15;
  const int kb = half * 8;
  const int bid = blockIdx.x;
  const int qt = bid & 127, bh = bid >> 7;
  const int q0 = qt * 16;
  const _Float16* Qp = Qh + (size_t)bh * SEQ * HD;
  const _Float16* Kp = Kh + (size_t)bh * SEQ * HD;
  const _Float16* Vp = Vt + (size_t)bh * HD * SEQ;

  v16h aq[2];
#pragma unroll
  for (int c = 0; c < 2; ++c) {
    const _Float16* base = Qp + (size_t)(q0 + l16) * HD + c * 32;
    aq[c] = cat8(*(const v8h*)(base + kb), *(const v8h*)(base + kb + 16));
  }
  v16h ones;
#pragma unroll
  for (int i = 0; i < 16; ++i) ones[i] = (_Float16)1.0f;

  float mi[8], li[8];
  v8f o[4] = {};
#pragma unroll
  for (int r = 0; r < 8; ++r) { mi[r] = -__builtin_inff(); li[r] = 0.f; }

  // prologue: DMA chunk 0 (K rows contiguous: 64-elem rows; V: 32-key rows)
  tdm_load_2d_f16(Kp, lds_off(&Ks[0][0][0]), HD, 32, HD, 4, 3);
  tdm_load_2d_f16(Vp, lds_off(&Vs[0][0][0]), 32, HD, SEQ, 3, 3);

  int t = 0;
  for (int k0 = 0; k0 <= q0 + 15; k0 += 32) {
    if (k0 + 32 <= q0 + 15) {           // prefetch next chunk into buf t^1
      tdm_load_2d_f16(Kp + (size_t)(k0 + 32) * HD,
                      lds_off(&Ks[t ^ 1][0][0]), HD, 32, HD, 4, 3);
      tdm_load_2d_f16(Vp + (size_t)(k0 + 32),
                      lds_off(&Vs[t ^ 1][0][0]), 32, HD, SEQ, 3, 3);
      __builtin_amdgcn_s_wait_tensorcnt(2);   // current (older) 2 ops done
    } else {
      __builtin_amdgcn_s_wait_tensorcnt(0);
    }

    v8f s[2] = {};
#pragma unroll
    for (int nt = 0; nt < 2; ++nt)
#pragma unroll
      for (int c = 0; c < 2; ++c) {
        const _Float16* kbase = &Ks[t][nt * 16 + l16][c * 32 + half * 8];
        v16h bk = cat8(*(const v8h*)kbase, *(const v8h*)(kbase + 16));
        s[nt] = wmma16(aq[c], bk, s[nt]);
      }
    if (k0 + 31 > q0) {                 // causal boundary tile
#pragma unroll
      for (int nt = 0; nt < 2; ++nt)
#pragma unroll
        for (int r = 0; r < 8; ++r) {
          int key = k0 + nt * 16 + l16;
          int qr = q0 + r + 8 * half;
          if (key > qr) s[nt][r] = -__builtin_inff();
        }
    }
    // running max (butterfly within 16-lane row halves) + p = exp(s - m)
    float sc[8];
#pragma unroll
    for (int r = 0; r < 8; ++r) {
      float v = fmaxf(s[0][r], s[1][r]);
#pragma unroll
      for (int off = 8; off >= 1; off >>= 1)
        v = fmaxf(v, __shfl_xor(v, off, 32));
      float mn = fmaxf(mi[r], v);
      sc[r] = __expf(mi[r] - mn);
      mi[r] = mn;
      s[0][r] = __expf(s[0][r] - mn);
      s[1][r] = __expf(s[1][r] - mn);
    }
    // C-layout -> A-layout transpose of P through LDS
#pragma unroll
    for (int nt = 0; nt < 2; ++nt)
#pragma unroll
      for (int r = 0; r < 8; ++r)
        P[r + 8 * half][nt * 16 + l16] = (_Float16)s[nt][r];
    __syncthreads();
    v16h pa = cat8(*(const v8h*)&P[l16][kb], *(const v8h*)&P[l16][kb + 16]);
    __syncthreads();

    // row-sum(P) = P @ ones : lands pre-broadcast in C layout
    v8f zero = {};
    v8f rs = wmma16(pa, ones, zero);
#pragma unroll
    for (int r = 0; r < 8; ++r) li[r] = li[r] * sc[r] + rs[r];
#pragma unroll
    for (int j = 0; j < 4; ++j)
#pragma unroll
      for (int r = 0; r < 8; ++r) o[j][r] *= sc[r];

#pragma unroll
    for (int nt = 0; nt < 4; ++nt) {
      const _Float16* vbase = &Vs[t][nt * 16 + l16][half * 8];
      v16h bv = cat8(*(const v8h*)vbase, *(const v8h*)(vbase + 16));
      o[nt] = wmma16(pa, bv, o[nt]);
    }
    t ^= 1;
  }

  const int b = bh >> 4, h = bh & 15;
#pragma unroll
  for (int nt = 0; nt < 4; ++nt)
#pragma unroll
    for (int r = 0; r < 8; ++r) {
      int qr = q0 + r + 8 * half;
      float val = o[nt][r] / li[r];
      Aout[((size_t)(b * SEQ + qr)) * D_EMBED + h * HD + nt * 16 + l16] =
          (_Float16)val;
    }
}

// ---------------------------------------------------------------------------
extern "C" void kernel_launch(void* const* d_in, const int* in_sizes, int n_in,
                              void* d_out, int out_size, void* d_ws,
                              size_t ws_size, hipStream_t stream) {
  (void)in_sizes; (void)n_in; (void)out_size; (void)ws_size;
  const float* x      = (const float*)d_in[0];
  const float* w_attn = (const float*)d_in[1];
  const float* b_attn = (const float*)d_in[2];
  const float* w_proj = (const float*)d_in[3];
  const float* b_proj = (const float*)d_in[4];
  float* out = (float*)d_out;

  char* ws = (char*)d_ws;
  const size_t MB = 1 << 20;
  _Float16* xh  = (_Float16*)(ws + 0 * MB);    // 8 MB  [4096][1024]
  _Float16* Wta = (_Float16*)(ws + 8 * MB);    // 6 MB  [3072][1024]
  _Float16* Wtp = (_Float16*)(ws + 14 * MB);   // 2 MB  [1024][1024]
  _Float16* Qh  = (_Float16*)(ws + 16 * MB);   // 8 MB
  _Float16* Kh  = (_Float16*)(ws + 24 * MB);   // 8 MB
  _Float16* Vt  = (_Float16*)(ws + 32 * MB);   // 8 MB
  _Float16* Ah  = xh;  // alias: xh dead after QKV GEMM, Ah written after it

  // 0) one-time f16 conversion / weight transposition
  cvt_f16<<<(MTOT * KDIM / 4) / 256, 256, 0, stream>>>(x, xh);
  transpose_cvt<<<dim3(3 * D_EMBED / 32, KDIM / 32), 256, 0, stream>>>(
      w_attn, Wta, KDIM, 3 * D_EMBED);
  transpose_cvt<<<dim3(D_EMBED / 32, KDIM / 32), 256, 0, stream>>>(
      w_proj, Wtp, KDIM, D_EMBED);
  // 1) qkv = x @ w_attn + b_attn -> head-major f16 Q/K/V (V transposed)
  gemm_wmma<0><<<dim3(MTOT / 64, 3072 / 128), 256, 0, stream>>>(
      xh, Wta, b_attn, 3 * D_EMBED, Qh, Kh, Vt, nullptr);
  // 2) causal flash attention
  attn_fa<<<BATCH * NHEAD * (SEQ / 16), 32, 0, stream>>>(Qh, Kh, Vt, Ah);
  // 3) out = attn @ w_proj + b_proj
  gemm_wmma<1><<<dim3(MTOT / 64, D_EMBED / 128), 256, 0, stream>>>(
      Ah, Wtp, b_proj, D_EMBED, nullptr, nullptr, nullptr, out);
}
